// TorchIndex_21818433864348
// MI455X (gfx1250) — matrix operand order repaired and checked by reading, not compiled, not run on hardware
//
#include <hip/hip_runtime.h>
#include <hip/hip_bf16.h>
#include <stdint.h>

// ---------------------------------------------------------------------------
// Top-k inner-product search, MI455X (gfx1250), wave32.
//   scores = Q(256x512) * V^T(512x500000), top-64 per row (sorted) + indices.
// Phase 1: bf16 WMMA GEMM (fp32->bf16 via v_perm_b32 truncation, A-frags
//          hoisted to registers) fused with exact per-chunk radix-select.
// Phase 2: merge 123 chunk-candidates per query -> sorted global top-64.
// ---------------------------------------------------------------------------

typedef __attribute__((ext_vector_type(16))) __bf16 v16bf;
typedef __attribute__((ext_vector_type(8)))  float  v8f;

constexpr int BQ     = 256;                      // queries
constexpr int HD     = 512;                      // hidden dim (K of GEMM)
constexpr int NV     = 500000;                   // database vectors
constexpr int KTOP   = 64;                       // top-k
constexpr int CHUNK  = 4096;                     // vectors per phase-1 block
constexpr int NCHUNK = (NV + CHUNK - 1) / CHUNK; // 123
constexpr int NCAND  = NCHUNK * KTOP;            // 7872 candidates / query

// fp32 -> bf16 (round to nearest even), low 16 bits (used once for the query)
__device__ __forceinline__ unsigned f2bf(float f) {
  unsigned u = __float_as_uint(f);
  return (u + 0x7FFFu + ((u >> 16) & 1u)) >> 16;
}
// pack bf16(hi):bf16(lo) from two floats in ONE v_perm_b32 (truncation)
__device__ __forceinline__ uint32_t pack_bf16_2(float lo, float hi) {
  return __builtin_amdgcn_perm(__float_as_uint(hi), __float_as_uint(lo),
                               0x07060302u);
}
// float -> monotonically ordered unsigned key (descending float == desc key)
__device__ __forceinline__ unsigned fkey(float f) {
  unsigned u = __float_as_uint(f);
  return (u & 0x80000000u) ? ~u : (u | 0x80000000u);
}

// ---------------------------------------------------------------------------
// Phase 1: one block = 16-query tile x 4096-vector chunk, 256 threads (8 waves)
// Dynamic LDS: 16KB query bf16 | 256KB chunk scores f32 | 8KB hist | 128B bcast
// ---------------------------------------------------------------------------
__global__ __launch_bounds__(256, 1) void topk_phase1(
    const float* __restrict__ query, const float* __restrict__ vectors,
    float* __restrict__ cand_s, int* __restrict__ cand_i)
{
  extern __shared__ char smem[];
  unsigned short* ldsQ = (unsigned short*)smem;                       // bf16 bits
  float*    ldsS = (float*)(smem + 16 * HD * 2);
  unsigned* ldsH = (unsigned*)(smem + 16 * HD * 2 + 16 * CHUNK * 4);
  unsigned* ldsX = ldsH + 8 * 256;

  const int tid  = threadIdx.x;
  const int wave = tid >> 5;
  const int lane = tid & 31;
  const int half = lane >> 4;   // wave32 half-select per WMMA VGPR layouts
  const int l16  = lane & 15;
  const int qtile = blockIdx.x; // 0..15  (fast dim -> L2 reuse of the chunk)
  const int chunk = blockIdx.y; // 0..122
  const long chunk_base = (long)chunk * CHUNK;

  // ---- query tile fp32 -> bf16 into LDS (once per block) --------------
  for (int i = tid; i < 16 * HD; i += 256) {
    int r = i >> 9, c = i & (HD - 1);
    ldsQ[i] = (unsigned short)f2bf(query[(qtile * 16 + r) * HD + c]);
  }
  __syncthreads();

  // ---- hoist ALL 16 A fragments (full K=512) into registers -----------
  // A 16x32 bf16 frag: elems 0-7 = K[kk+half*8..+8), elems 8-15 = +16.
  const uint32_t* qrow = (const uint32_t*)(ldsQ + l16 * HD);
  uint32_t Ad[16][8];
#pragma unroll
  for (int f = 0; f < 16; ++f) {
    const int kk = f * 32;
    const int o0 = (kk + half * 8) >> 1;
    const int o1 = (kk + 16 + half * 8) >> 1;
#pragma unroll
    for (int j = 0; j < 4; ++j) { Ad[f][j] = qrow[o0 + j]; Ad[f][4 + j] = qrow[o1 + j]; }
  }

  // ---- GEMM: each wave owns columns [wave*512, wave*512+512) ----------
  for (int t = 0; t < 32; ++t) {
    const int  ncol0 = wave * 512 + t * 16;
    const long n  = chunk_base + ncol0 + l16;
    const long nc = (n < NV) ? n : (long)(NV - 1);          // clamp tail loads
    const float* vp = vectors + nc * HD + half * 8;         // B frag row base
    __builtin_prefetch(vp + 16 * HD, 0, 0);                 // next tile stream

    v8f acc = {};
#pragma unroll
    for (int f = 0; f < 16; ++f) {
      // B fragment: 2x8 fp32 (K +0..8, +16..24) -> 8 packed bf16 dwords (8 v_perm)
      const float* bp = vp + f * 32;
      union { uint32_t d[8]; v16bf v; } Bu;
#pragma unroll
      for (int j = 0; j < 8; ++j) Bu.d[j] = pack_bf16_2(bp[2 * j + (j >> 2) * 8], bp[2 * j + 1 + (j >> 2) * 8]);

      union { uint32_t d[8]; v16bf v; } Au;
#pragma unroll
      for (int j = 0; j < 8; ++j) Au.d[j] = Ad[f][j];

      acc = __builtin_amdgcn_wmma_f32_16x16x32_bf16(
          false, Au.v, false, Bu.v, (short)0, acc, false, false);
    }
    // C layout: VGPR i -> row i + half*8, col = l16
#pragma unroll
    for (int i = 0; i < 8; ++i) {
      const int row = i + half * 8;
      ldsS[row * CHUNK + ncol0 + l16] = (n < NV) ? acc[i] : -__builtin_inff();
    }
  }
  __syncthreads();

  // ---- exact per-row top-64 via 4-pass MSB radix select ---------------
  unsigned* hist = ldsH + wave * 256;
  unsigned* xw   = ldsX + wave * 4;

  for (int rr = 0; rr < 2; ++rr) {                 // 8 waves x 2 rows = 16 rows
    const int row = wave * 2 + rr;
    const float* srow = ldsS + row * CHUNK;
    unsigned prefix = 0, pmask = 0;
    int need = KTOP;

    for (int pass = 0; pass < 4; ++pass) {
      const int shift = 24 - pass * 8;
      for (int i = lane; i < 256; i += 32) hist[i] = 0u;
      __syncthreads();
      for (int idx = lane; idx < CHUNK; idx += 32) {
        unsigned key = fkey(srow[idx]);
        if ((key & pmask) == prefix) atomicAdd(&hist[(key >> shift) & 255u], 1u);
      }
      __syncthreads();
      if (lane == 0) {
        unsigned cum = 0; int b = 255;
        for (;; --b) { cum += hist[b]; if ((int)cum >= need || b == 0) break; }
        xw[0] = (unsigned)b;
        xw[1] = cum - hist[b];                     // count strictly above bin b
      }
      __syncthreads();
      const unsigned b = xw[0];
      need  -= (int)xw[1];
      prefix |= b << shift;
      pmask  |= 0xFFu << shift;
      __syncthreads();
    }

    // prefix == exact key of the 64th-largest; need == #ties still wanted
    const unsigned T = prefix;
    if (lane == 0) { xw[1] = 0u; xw[2] = 0u; }
    __syncthreads();
    const int q = qtile * 16 + row;
    float* cs = cand_s + (size_t)q * NCAND + (size_t)chunk * KTOP;
    int*   ci = cand_i + (size_t)q * NCAND + (size_t)chunk * KTOP;
    const int gtotal = KTOP - need;                // exactly count(key > T)
    for (int idx = lane; idx < CHUNK; idx += 32) {
      float s = srow[idx];
      unsigned key = fkey(s);
      if (key > T) {
        unsigned pos = atomicAdd(&xw[1], 1u);
        cs[pos] = s; ci[pos] = (int)(chunk_base + idx);
      } else if (key == T) {
        unsigned pos = atomicAdd(&xw[2], 1u);
        if ((int)pos < need) {
          cs[gtotal + pos] = s; ci[gtotal + pos] = (int)(chunk_base + idx);
        }
      }
    }
    __syncthreads();
  }
}

// ---------------------------------------------------------------------------
// Phase 2: one block per query; merge 7872 candidates -> sorted top-64.
// ---------------------------------------------------------------------------
__global__ __launch_bounds__(256, 1) void topk_phase2(
    const float* __restrict__ cand_s, const int* __restrict__ cand_i,
    float* __restrict__ out)
{
  __shared__ float s[NCAND];     // ~31.5 KB
  __shared__ float rs[256];
  __shared__ int   rp[256];

  const int q = blockIdx.x, tid = threadIdx.x;
  const float* cs = cand_s + (size_t)q * NCAND;
  const int*   ci = cand_i + (size_t)q * NCAND;
  for (int i = tid; i < NCAND; i += 256) s[i] = cs[i];
  __syncthreads();

  float* outS = out + (size_t)q * KTOP;
  float* outI = out + (size_t)BQ * KTOP + (size_t)q * KTOP;

  for (int k = 0; k < KTOP; ++k) {
    float best = -__builtin_inff(); int bp = 0;
    for (int i = tid; i < NCAND; i += 256)
      if (s[i] > best) { best = s[i]; bp = i; }
    rs[tid] = best; rp[tid] = bp;
    __syncthreads();
    for (int off = 128; off > 0; off >>= 1) {
      if (tid < off && rs[tid + off] > rs[tid]) {
        rs[tid] = rs[tid + off]; rp[tid] = rp[tid + off];
      }
      __syncthreads();
    }
    if (tid == 0) {
      int p = rp[0];
      outS[k] = rs[0];
      outI[k] = (float)ci[p];    // indices value-preserved as float (< 2^24)
      s[p] = -__builtin_inff();  // remove winner
    }
    __syncthreads();
  }
}

// ---------------------------------------------------------------------------
extern "C" void kernel_launch(void* const* d_in, const int* in_sizes, int n_in,
                              void* d_out, int out_size, void* d_ws, size_t ws_size,
                              hipStream_t stream) {
  (void)in_sizes; (void)n_in; (void)out_size; (void)ws_size;
  const float* query   = (const float*)d_in[0];
  const float* vectors = (const float*)d_in[1];
  // d_in[2] is k == 64, baked in at compile time.

  float* cand_s = (float*)d_ws;                                  // 8.06 MB
  int*   cand_i = (int*)((char*)d_ws +
                         (size_t)BQ * NCAND * sizeof(float));    // 8.06 MB

  const size_t lds1 = (size_t)16 * HD * 2       // query tile bf16
                    + (size_t)16 * CHUNK * 4    // chunk scores f32
                    + (size_t)8 * 256 * 4       // per-wave histograms
                    + (size_t)8 * 4 * 4;        // per-wave broadcast slots

  dim3 g1(BQ / 16, NCHUNK);   // x = query tile (fast) -> chunk stays in L2
  topk_phase1<<<g1, 256, lds1, stream>>>(query, vectors, cand_s, cand_i);
  topk_phase2<<<BQ, 256, 0, stream>>>(cand_s, cand_i, (float*)d_out);
}
